// MHA_13443247636500
// MI455X (gfx1250) — matrix operand order repaired and checked by reading, not compiled, not run on hardware
//
#include <hip/hip_runtime.h>
#include <math.h>

// ---------------------------------------------------------------------------
// MHA forward for MI455X (gfx1250, wave32).
// Roofline: attn output alone = 536MB fp32 -> >=23us of pure stores at
// 23.3 TB/s; total math ~70 GFLOP (trivial for f16 WMMA). Strategy:
//  * f16 WMMA (16x16x32) with fp32 accumulation everywhere
//  * attn written exactly once, LDS-staged float4 NON-TEMPORAL stores
//    (write-once data must not evict the L2-resident Q/K/V projections)
//  * Qp/Kp/Vp (48MB) + ctx (16MB) fit in the 192MB L2 -> regular TH, each
//    tile re-read by 32 workgroups hits L2 instead of HBM
//  * pass-2 recomputes QK^T (+17 GF) instead of re-reading 536MB of scores
//  * GEMM: double-buffered LDS, 1 barrier/K-step, loads overlap WMMA
//  * global_prefetch of next K/V tile inside the attention loop
// ---------------------------------------------------------------------------

typedef __attribute__((ext_vector_type(16))) _Float16 v16h;
typedef __attribute__((ext_vector_type(8)))  _Float16 v8h;
typedef __attribute__((ext_vector_type(8)))  float    v8f;
typedef __attribute__((ext_vector_type(4)))  float    f4v;

#define DMODEL 1024
#define NHEAD  16
#define DK     64
#define SEQ    2048
#define BATCH  2
#define MTOT   (BATCH*SEQ)   // 4096

static __device__ __forceinline__ v8f wmma_f16(v16h a, v16h b, v8f c) {
  // D = A(16x32 f16) * B(32x16 f16) + C(16x16 f32)
  return __builtin_amdgcn_wmma_f32_16x16x32_f16(
      /*neg_a=*/false, a, /*neg_b=*/false, b,
      /*c_mod=*/(short)0, c, /*reuse_a=*/false, /*reuse_b=*/false);
}

// ---------------------------------------------------------------------------
// Y[M=4096, N=1024] = X[M, K=1024] @ W[N, K]^T + bias[N]
// fp32 in/out, f16 WMMA compute. WG tile 128x64, 8 waves, wave tile 32x32.
// Double-buffered LDS: stage K-step kt+1 while WMMAs consume kt.
// NT=true -> non-temporal epilogue stores (output never re-read).
// ---------------------------------------------------------------------------
template<bool NT>
__global__ __launch_bounds__(256)
void gemm_xwt_bias_kernel(const float* __restrict__ X, const float* __restrict__ W,
                          const float* __restrict__ bias, float* __restrict__ Y)
{
  const int K = DMODEL, N = DMODEL;
  __shared__ _Float16 As[2][128][32];   // [buf][m][k]
  __shared__ _Float16 Bs[2][32][64];    // [buf][k][n] == W[n][k] transposed

  const int t = threadIdx.x;
  const int lane = t & 31, wid = t >> 5;
  const int m0 = blockIdx.y * 128;
  const int n0 = blockIdx.x * 64;
  const int wm = (wid & 3) * 32;     // wave row base in tile
  const int wn = (wid >> 2) * 32;    // wave col base in tile
  const int kh = (lane >> 4) * 8;    // A-fragment K-half select

  v8f acc[2][2] = {};

  auto stage = [&](int buf, int k0) {
#pragma unroll
    for (int i = 0; i < 4; ++i) {
      int lin = t + 256 * i;                  // 1024 float4 slots
      int r = lin >> 3, c = (lin & 7) << 2;
      float4 av = *(const float4*)(X + (size_t)(m0 + r) * K + k0 + c);
      As[buf][r][c+0] = (_Float16)av.x;  As[buf][r][c+1] = (_Float16)av.y;
      As[buf][r][c+2] = (_Float16)av.z;  As[buf][r][c+3] = (_Float16)av.w;
    }
#pragma unroll
    for (int i = 0; i < 2; ++i) {
      int lin = t + 256 * i;                  // 512 float4 slots
      int nr = lin >> 3, c = (lin & 7) << 2;
      float4 wv = *(const float4*)(W + (size_t)(n0 + nr) * K + k0 + c);
      Bs[buf][c+0][nr] = (_Float16)wv.x;  Bs[buf][c+1][nr] = (_Float16)wv.y;
      Bs[buf][c+2][nr] = (_Float16)wv.z;  Bs[buf][c+3][nr] = (_Float16)wv.w;
    }
  };

  stage(0, 0);
  __syncthreads();

  const int NK = K / 32;
  for (int kt = 0; kt < NK; ++kt) {
    const int cur = kt & 1;
    if (kt + 1 < NK) stage(cur ^ 1, (kt + 1) * 32);   // overlap with WMMA below

#pragma unroll
    for (int mi = 0; mi < 2; ++mi) {
      v16h a;
      ((v8h*)&a)[0] = *(const v8h*)&As[cur][wm + mi*16 + (lane & 15)][kh];
      ((v8h*)&a)[1] = *(const v8h*)&As[cur][wm + mi*16 + (lane & 15)][kh + 16];
#pragma unroll
      for (int ni = 0; ni < 2; ++ni) {
        v16h b;   // lane holds B[k=kh+s, kh+16+s][n=lane%16]
#pragma unroll
        for (int s = 0; s < 8; ++s) {
          b[s]     = Bs[cur][kh + s][wn + ni*16 + (lane & 15)];
          b[s + 8] = Bs[cur][kh + 16 + s][wn + ni*16 + (lane & 15)];
        }
        acc[mi][ni] = wmma_f16(a, b, acc[mi][ni]);
      }
    }
    __syncthreads();   // readers done with cur before it is restaged next iter
  }

  // epilogue: C frag layout -> row = base + r + 8*(lane/16), col = base + lane%16
  const int cn = lane & 15;
  const int rh = (lane >> 4) * 8;
#pragma unroll
  for (int mi = 0; mi < 2; ++mi) {
#pragma unroll
    for (int ni = 0; ni < 2; ++ni) {
      int col = n0 + wn + ni*16 + cn;
      float bv = bias[col];
      float* yp = Y + (size_t)(m0 + wm + mi*16 + rh) * N + col;
#pragma unroll
      for (int r = 0; r < 8; ++r) {
        float val = acc[mi][ni][r] + bv;
        if (NT) __builtin_nontemporal_store(val, yp + (size_t)r * N);
        else    yp[(size_t)r * N] = val;
      }
    }
  }
}

// ---------------------------------------------------------------------------
// Attention per (b,h): flash-style online softmax + P@V in pass 1 (-> ctx),
// score recompute in pass 2 (-> normalized attn, LDS-staged NT float4 stores).
// WG handles 64 query rows; loops over 32 key tiles of 64.
// ---------------------------------------------------------------------------
__global__ __launch_bounds__(256)
void mha_attn_kernel(const float* __restrict__ Qp, const float* __restrict__ Kp,
                     const float* __restrict__ Vp, float* __restrict__ ctx,
                     float* __restrict__ attn)
{
  __shared__ _Float16 Qs [64][64];   // [q][d], pre-scaled by 1/sqrt(Dk)
  __shared__ _Float16 Kst[64][64];   // [d][key]  (transposed for B-fragments)
  __shared__ _Float16 Vs [64][64];   // [key][d]
  __shared__ _Float16 Ps [64][64];   // exp(s - m) probs, f16
  __shared__ float    Ss [64][68];   // f32 tile: scores (pass1) / attn staging (pass2)
  __shared__ float    mrow[64], lrow[64], fsc[64];

  const int t = threadIdx.x, lane = t & 31, wid = t >> 5;
  const int qb = blockIdx.x;                 // 0..31 (64-row query block)
  const int bh = blockIdx.y;                 // b*NHEAD + h
  const int b = bh >> 4, h = bh & 15;
  const int q0 = qb * 64;
  const float scale = 0.125f;                // 1/sqrt(64)

  // stage Q tile once, scale folded in
#pragma unroll
  for (int i = 0; i < 4; ++i) {
    int lin = t + 256*i;
    int r = lin >> 4, c = (lin & 15) << 2;
    float4 qv = *(const float4*)(Qp + (size_t)(b*SEQ + q0 + r)*DMODEL + h*DK + c);
    Qs[r][c+0] = (_Float16)(qv.x * scale);  Qs[r][c+1] = (_Float16)(qv.y * scale);
    Qs[r][c+2] = (_Float16)(qv.z * scale);  Qs[r][c+3] = (_Float16)(qv.w * scale);
  }
  if (t < 64) { mrow[t] = -3.0e38f; lrow[t] = 0.0f; }
  __syncthreads();

  const int qr    = (wid & 3) * 16;          // wave q-row base
  const int wc    = (wid >> 2) * 32;         // wave col base (key or dk)
  const int kh    = (lane >> 4) * 8;
  const int am    = qr + (lane & 15);
  const int rbase = qr + (lane >> 4) * 8;

  v8f oacc[2] = {};                          // ctx accum: 16 rows x 32 dk

  // ---------------- pass 1: online softmax + P@V ----------------
  for (int kb = 0; kb < 32; ++kb) {
    // prefetch next K/V tile (L2-resident; pull toward WGP early)
    if (kb + 1 < 32) {
      int pi = t & 127;
      int r = pi >> 1, half = (pi & 1) << 5;          // 2x128B per 64-float row
      const float* base = (t < 128) ? Kp : Vp;
      __builtin_prefetch(base + (size_t)(b*SEQ + (kb+1)*64 + r)*DMODEL + h*DK + half, 0, 3);
    }
#pragma unroll
    for (int i = 0; i < 4; ++i) {
      int lin = t + 256*i;
      int r = lin >> 4, c = (lin & 15) << 2;
      size_t off = (size_t)(b*SEQ + kb*64 + r)*DMODEL + h*DK + c;
      float4 kv = *(const float4*)(Kp + off);
      float4 vv = *(const float4*)(Vp + off);
      Kst[c+0][r] = (_Float16)kv.x;  Kst[c+1][r] = (_Float16)kv.y;
      Kst[c+2][r] = (_Float16)kv.z;  Kst[c+3][r] = (_Float16)kv.w;
      Vs[r][c+0] = (_Float16)vv.x;   Vs[r][c+1] = (_Float16)vv.y;
      Vs[r][c+2] = (_Float16)vv.z;   Vs[r][c+3] = (_Float16)vv.w;
    }
    __syncthreads();

    // scores: S = Q @ K^T (contraction over d=64, two WMMA k-steps)
    v8f sacc[2] = {};
#pragma unroll
    for (int d = 0; d < 64; d += 32) {
      v16h a;
      ((v8h*)&a)[0] = *(const v8h*)&Qs[am][d + kh];
      ((v8h*)&a)[1] = *(const v8h*)&Qs[am][d + kh + 16];
#pragma unroll
      for (int ci = 0; ci < 2; ++ci) {
        v16h bm;  // lane holds K^T[d=d+kh+s, d+kh+16+s][key=lane%16]
#pragma unroll
        for (int s = 0; s < 8; ++s) {
          bm[s]     = Kst[d + kh + s][wc + ci*16 + (lane & 15)];
          bm[s + 8] = Kst[d + kh + 16 + s][wc + ci*16 + (lane & 15)];
        }
        sacc[ci] = wmma_f16(a, bm, sacc[ci]);
      }
    }
    // dump scores to LDS for row-wise softmax
#pragma unroll
    for (int ci = 0; ci < 2; ++ci) {
      int col = wc + ci*16 + (lane & 15);
#pragma unroll
      for (int r = 0; r < 8; ++r)
        Ss[rbase + r][col] = sacc[ci][r];
    }
    __syncthreads();

    // online softmax update (one thread per row)
    if (t < 64) {
      float mo = mrow[t], mn = mo;
#pragma unroll 8
      for (int j = 0; j < 64; ++j) mn = fmaxf(mn, Ss[t][j]);
      float f = __expf(mo - mn);
      float s = 0.0f;
#pragma unroll 8
      for (int j = 0; j < 64; ++j) {
        float e = __expf(Ss[t][j] - mn);
        s += e;
        Ps[t][j] = (_Float16)e;
      }
      lrow[t] = lrow[t] * f + s;
      mrow[t] = mn;
      fsc[t]  = f;
    }
    __syncthreads();

    // rescale running ctx, then ctx += P @ V
#pragma unroll
    for (int r = 0; r < 8; ++r) {
      float f = fsc[rbase + r];
      oacc[0][r] *= f;
      oacc[1][r] *= f;
    }
#pragma unroll
    for (int kt = 0; kt < 64; kt += 32) {
      v16h a;
      ((v8h*)&a)[0] = *(const v8h*)&Ps[am][kt + kh];
      ((v8h*)&a)[1] = *(const v8h*)&Ps[am][kt + kh + 16];
#pragma unroll
      for (int ci = 0; ci < 2; ++ci) {
        v16h bm;  // lane holds V[key=kt+kh+s, kt+kh+16+s][dk=lane%16]
#pragma unroll
        for (int s = 0; s < 8; ++s) {
          bm[s]     = Vs[kt + kh + s][wc + ci*16 + (lane & 15)];
          bm[s + 8] = Vs[kt + kh + 16 + s][wc + ci*16 + (lane & 15)];
        }
        oacc[ci] = wmma_f16(a, bm, oacc[ci]);
      }
    }
    __syncthreads();
  }

  // write normalized ctx in [B*S, DMODEL] layout (re-read by out-proj: keep in L2)
#pragma unroll
  for (int ci = 0; ci < 2; ++ci) {
    int col = h*DK + wc + ci*16 + (lane & 15);
#pragma unroll
    for (int r = 0; r < 8; ++r) {
      float inv = 1.0f / lrow[rbase + r];
      ctx[(size_t)(b*SEQ + q0 + rbase + r)*DMODEL + col] = oacc[ci][r] * inv;
    }
  }
  __syncthreads();

  // ---------------- pass 2: recompute scores, write attn once ----------------
  for (int kb = 0; kb < 32; ++kb) {
    if (kb + 1 < 32 && t < 128) {
      int r = t >> 1, half = (t & 1) << 5;
      __builtin_prefetch(Kp + (size_t)(b*SEQ + (kb+1)*64 + r)*DMODEL + h*DK + half, 0, 3);
    }
#pragma unroll
    for (int i = 0; i < 4; ++i) {
      int lin = t + 256*i;
      int r = lin >> 4, c = (lin & 15) << 2;
      float4 kv = *(const float4*)(Kp + (size_t)(b*SEQ + kb*64 + r)*DMODEL + h*DK + c);
      Kst[c+0][r] = (_Float16)kv.x;  Kst[c+1][r] = (_Float16)kv.y;
      Kst[c+2][r] = (_Float16)kv.z;  Kst[c+3][r] = (_Float16)kv.w;
    }
    __syncthreads();

    v8f sacc[2] = {};
#pragma unroll
    for (int d = 0; d < 64; d += 32) {
      v16h a;
      ((v8h*)&a)[0] = *(const v8h*)&Qs[am][d + kh];
      ((v8h*)&a)[1] = *(const v8h*)&Qs[am][d + kh + 16];
#pragma unroll
      for (int ci = 0; ci < 2; ++ci) {
        v16h bm;
#pragma unroll
        for (int s = 0; s < 8; ++s) {
          bm[s]     = Kst[d + kh + s][wc + ci*16 + (lane & 15)];
          bm[s + 8] = Kst[d + kh + 16 + s][wc + ci*16 + (lane & 15)];
        }
        sacc[ci] = wmma_f16(a, bm, sacc[ci]);
      }
    }
    // normalize into LDS staging tile
#pragma unroll
    for (int ci = 0; ci < 2; ++ci) {
      int col = wc + ci*16 + (lane & 15);
#pragma unroll
      for (int r = 0; r < 8; ++r) {
        int row = rbase + r;
        Ss[row][col] = __expf(sacc[ci][r] - mrow[row]) * (1.0f / lrow[row]);
      }
    }
    __syncthreads();

    // coalesced non-temporal float4 stores of the 64x64 attn tile
#pragma unroll
    for (int i = 0; i < 4; ++i) {
      int lin = t + 256*i;
      int r = lin >> 4, c = (lin & 15) << 2;
      f4v val = *(const f4v*)&Ss[r][c];
      __builtin_nontemporal_store(
          val, (f4v*)(attn + ((size_t)bh*SEQ + q0 + r)*SEQ + kb*64 + c));
    }
    __syncthreads();
  }
}

// ---------------------------------------------------------------------------
extern "C" void kernel_launch(void* const* d_in, const int* in_sizes, int n_in,
                              void* d_out, int out_size, void* d_ws, size_t ws_size,
                              hipStream_t stream) {
  (void)in_sizes; (void)n_in; (void)out_size; (void)ws_size;
  const float* q  = (const float*)d_in[0];
  const float* k  = (const float*)d_in[1];
  const float* v  = (const float*)d_in[2];
  const float* Wq = (const float*)d_in[3];
  const float* bq = (const float*)d_in[4];
  const float* Wk = (const float*)d_in[5];
  const float* bk = (const float*)d_in[6];
  const float* Wv = (const float*)d_in[7];
  const float* bv = (const float*)d_in[8];
  const float* Wo = (const float*)d_in[9];
  const float* bo = (const float*)d_in[10];

  float* out  = (float*)d_out;                         // [B*S, DMODEL]
  float* attn = out + (size_t)MTOT * DMODEL;           // [B*H, S, S]

  const size_t proj = (size_t)MTOT * DMODEL;           // 4M floats each
  float* Qp  = (float*)d_ws;
  float* Kp  = Qp + proj;
  float* Vp  = Kp + proj;
  float* ctx = Vp + proj;                              // 64MB total scratch

  dim3 gblk(256);
  dim3 ggrid(DMODEL / 64, MTOT / 128);                 // (16, 32)
  // Q/K/V/ctx are re-read -> regular temporal stores (L2-resident)
  gemm_xwt_bias_kernel<false><<<ggrid, gblk, 0, stream>>>(q, Wq, bq, Qp);
  gemm_xwt_bias_kernel<false><<<ggrid, gblk, 0, stream>>>(k, Wk, bk, Kp);
  gemm_xwt_bias_kernel<false><<<ggrid, gblk, 0, stream>>>(v, Wv, bv, Vp);

  dim3 agrid(SEQ / 64, BATCH * NHEAD);                 // (32, 32)
  mha_attn_kernel<<<agrid, gblk, 0, stream>>>(Qp, Kp, Vp, ctx, attn);

  // final output never re-read -> non-temporal epilogue
  gemm_xwt_bias_kernel<true><<<ggrid, gblk, 0, stream>>>(ctx, Wo, bo, out);
}
